// Net_48180943126545
// MI455X (gfx1250) — matrix operand order repaired and checked by reading, not compiled, not run on hardware
//
#include <hip/hip_runtime.h>

// CDNA5 (gfx1250) wave32 WMMA neural-ODE RK4 kernel.
// Layer1: V_WMMA_F32_16X16X4_F32  (H^T = W1 * Y^T + b1), f32 precision.
// Layer2: V_WMMA_F32_16X16X32_F16 (O^T = W2 * tanh(H)^T + b2).
// All operand layout conversions are per-lane packs, a ds_bpermute and SWAPX16 ds_swizzles.

typedef __attribute__((ext_vector_type(2)))  float    v2f;
typedef __attribute__((ext_vector_type(8)))  float    v8f;
typedef __attribute__((ext_vector_type(16))) _Float16 v16h;

__device__ __forceinline__ float dtanh_f32(float x) {
#if __has_builtin(__builtin_amdgcn_tanhf)
  return __builtin_amdgcn_tanhf(x);      // v_tanh_f32 (gfx1250 TRANS op)
#else
  return ::tanhf(x);
#endif
}

__device__ __forceinline__ unsigned pack_f16x2(float a, float b) {
#if __has_builtin(__builtin_amdgcn_cvt_pkrtz)
  auto p = __builtin_amdgcn_cvt_pkrtz(a, b);   // v_cvt_pk_rtz_f16_f32
  return __builtin_bit_cast(unsigned, p);
#else
  unsigned l = __builtin_bit_cast(unsigned short, (_Float16)a);
  unsigned h = __builtin_bit_cast(unsigned short, (_Float16)b);
  return l | (h << 16);
#endif
}

__global__ __launch_bounds__(256) void node_rk4_wmma(
    const float* __restrict__ x,
    const float* __restrict__ W1, const float* __restrict__ b1,
    const float* __restrict__ W2, const float* __restrict__ b2,
    const float* __restrict__ t1,
    float* __restrict__ out, int n) {
  // t_seq = [0, .01, .02, .03, .04, t1]
  if (blockIdx.x == 0 && threadIdx.x < 6)
    out[threadIdx.x] = (threadIdx.x < 5) ? (float)(0.01 * (double)threadIdx.x)
                                         : t1[0];

  const int lane   = threadIdx.x & 31;
  const int wave   = threadIdx.x >> 5;
  const int ntiles = (n + 15) >> 4;
  const int tile   = blockIdx.x * 8 + wave;      // 16 points per wave
  if (tile >= ntiles) return;                    // wave-uniform exit

  const bool lo16 = lane < 16;
  const int  m    = lane & 15;                   // point / row index
  const int  hb   = (lane >> 4) << 3;            // 0 or 8 (C/D row-half)

  // ---- constant operands (built once per wave) ----------------------------
  // Layer1 A tiles: W1 rows (32x3 -> two 16x4 f32 A tiles, K=3 used).
  // f32 A 16x4 layout: VGPR0 = K0(lanes0-15)/K2(lanes16-31), VGPR1 = K1/K3.
  v2f A1[2];
  v8f C1[2];                                     // bias b1 as the C operand
#pragma unroll
  for (int t = 0; t < 2; ++t) {
    const int row = t * 16 + m;
    A1[t].x = lo16 ? W1[row * 3 + 0] : W1[row * 3 + 2];
    A1[t].y = lo16 ? W1[row * 3 + 1] : 0.0f;     // K=3 pad
#pragma unroll
    for (int j = 0; j < 8; ++j) C1[t][j] = b1[t * 16 + j + hb];
  }
  // Layer2 A: W2 (3x32) zero-padded to 16x32, f16, contiguous k order used here:
  // lanes0-15: elems0-15 -> K0-15; lanes16-31: elems0-15 -> K16-31.
  v16h A2;
#pragma unroll
  for (int i = 0; i < 16; ++i) {
    const int k = i + (hb << 1);
    A2[i] = (m < 3) ? (_Float16)W2[m * 32 + k] : (_Float16)0.0f;
  }
  v8f C2;                                        // bias b2 (rows 3..15 = 0)
#pragma unroll
  for (int j = 0; j < 8; ++j) {
    const int od = j + hb;
    C2[j] = (od < 3) ? b2[od] : 0.0f;
  }

  // ---- one MLP evaluation: k = W2*tanh(W1*y + b1) + b2 --------------------
  auto mlp = [&](float s0, float s1, float s2,
                 float& kx, float& ky, float& kz) {
    // B1 (f32 4x16 B-layout): lanes0-15 {y0,y1}, lanes16-31 {y2,0}.
    const float s2b = __int_as_float(
        __builtin_amdgcn_ds_bpermute(m << 2, __float_as_int(s2)));
    v2f B1v;
    B1v.x = lo16 ? s0 : s2b;
    B1v.y = lo16 ? s1 : 0.0f;
    // H^T tiles (hidden 0-15 and 16-31) x 16 points, f32 accumulate.
    v8f H0 = __builtin_amdgcn_wmma_f32_16x16x4_f32(
        false, A1[0], false, B1v, (short)0, C1[0], false, false);
    v8f H1 = __builtin_amdgcn_wmma_f32_16x16x4_f32(
        false, A1[1], false, B1v, (short)0, C1[1], false, false);
    // tanh + pack to f16 pairs (per-lane; lane l holds point l%16's hidden).
    unsigned p0[4], p1[4];
#pragma unroll
    for (int j = 0; j < 4; ++j) {
      p0[j] = pack_f16x2(dtanh_f32(H0[2 * j]), dtanh_f32(H0[2 * j + 1]));
      p1[j] = pack_f16x2(dtanh_f32(H1[2 * j]), dtanh_f32(H1[2 * j + 1]));
    }
    // f16 B uses A2's k order: lanes0-15 = K0-15, lanes16-31 = K16-31; the C-layout
    // halves are split {0-7|8-15} per tile -> swap halves with SWAPX16.
    union { v16h h; unsigned u[8]; } B2;
#pragma unroll
    for (int j = 0; j < 4; ++j) {
      unsigned q0 = (unsigned)__builtin_amdgcn_ds_swizzle((int)p0[j], 0x401F);
      unsigned q1 = (unsigned)__builtin_amdgcn_ds_swizzle((int)p1[j], 0x401F);
      B2.u[j]     = lo16 ? p0[j] : q1;   // K 0-7  | K 16-23
      B2.u[4 + j] = lo16 ? q0    : p1[j];// K 8-15 | K 24-31
    }
    v8f O = __builtin_amdgcn_wmma_f32_16x16x32_f16(
        false, A2, false, B2.h, (short)0, C2, false, false);
    kx = O[0]; ky = O[1]; kz = O[2];     // output dims 0..2, lane = point
  };

  // ---- state + RK4 (3/8 rule), y lives on lanes 0-15 ----------------------
  const int  p   = tile * 16 + m;
  const bool act = lo16 && (p < n);
  float y0 = 0.0f, y1 = 0.0f, y2 = 0.0f;
  if (act) {
    y0 = x[p * 3 + 0]; y1 = x[p * 3 + 1]; y2 = x[p * 3 + 2];
  }
  float* yo = out + 6;
  if (act) {  // y[0] = x
    yo[(size_t)p * 3 + 0] = y0;
    yo[(size_t)p * 3 + 1] = y1;
    yo[(size_t)p * 3 + 2] = y2;
  }
  const float tf = t1[0];
  const float c3 = 1.0f / 3.0f;
#pragma unroll
  for (int s = 0; s < 5; ++s) {
    const float h = (s < 4) ? 0.01f : (tf - 0.04f);
    float k1x, k1y, k1z, k2x, k2y, k2z, k3x, k3y, k3z, k4x, k4y, k4z;
    mlp(y0, y1, y2, k1x, k1y, k1z);
    mlp(y0 + h * c3 * k1x, y1 + h * c3 * k1y, y2 + h * c3 * k1z,
        k2x, k2y, k2z);
    mlp(y0 + h * (k2x - c3 * k1x), y1 + h * (k2y - c3 * k1y),
        y2 + h * (k2z - c3 * k1z), k3x, k3y, k3z);
    mlp(y0 + h * (k1x - k2x + k3x), y1 + h * (k1y - k2y + k3y),
        y2 + h * (k1z - k2z + k3z), k4x, k4y, k4z);
    y0 += h * 0.125f * (k1x + 3.0f * (k2x + k3x) + k4x);
    y1 += h * 0.125f * (k1y + 3.0f * (k2y + k3y) + k4y);
    y2 += h * 0.125f * (k1z + 3.0f * (k2z + k3z) + k4z);
    if (act) {
      const size_t off = (size_t)(s + 1) * (size_t)n * 3 + (size_t)p * 3;
      yo[off + 0] = y0; yo[off + 1] = y1; yo[off + 2] = y2;
    }
  }
}

extern "C" void kernel_launch(void* const* d_in, const int* in_sizes, int n_in,
                              void* d_out, int out_size, void* d_ws,
                              size_t ws_size, hipStream_t stream) {
  const float* x  = (const float*)d_in[0];
  const float* W1 = (const float*)d_in[1];
  const float* b1 = (const float*)d_in[2];
  const float* W2 = (const float*)d_in[3];
  const float* b2 = (const float*)d_in[4];
  const float* t1 = (const float*)d_in[5];
  float* out = (float*)d_out;
  const int n      = in_sizes[0] / 3;          // 1,000,000 points
  const int ntiles = (n + 15) / 16;
  const int blocks = (ntiles + 7) / 8;         // 8 waves (tiles) per block
  node_rk4_wmma<<<dim3(blocks), dim3(256), 0, stream>>>(
      x, W1, b1, W2, b2, t1, out, n);
}
